// AttentionWeightNet_50525995270370
// MI455X (gfx1250) — matrix operand order repaired and checked by reading, not compiled, not run on hardware
//
#include <hip/hip_runtime.h>
#include <hip/hip_bf16.h>
#include <math.h>

// ---------------------------------------------------------------------------
// Types for CDNA5 WMMA (wave32): A/B = 16 bf16 per lane (8 VGPRs),
// C/D = 8 f32 per lane (8 VGPRs).
// ---------------------------------------------------------------------------
typedef __attribute__((ext_vector_type(16))) __bf16 bf16x16;
typedef __attribute__((ext_vector_type(8)))  float  f32x8;

static_assert(sizeof(__bf16) == 2, "bf16 size");

__device__ __forceinline__ f32x8 wmma_bf16(const bf16x16& a, const bf16x16& b,
                                           const f32x8& c) {
  // (neg_a, A, neg_b, B, c_mod, C, reuse_a, reuse_b)
  return __builtin_amdgcn_wmma_f32_16x16x32_bf16(false, a, false, b,
                                                 (short)0, c, false, false);
}

// A fragment (16x32, row-major source, stride ld in bf16 elements).
// Contiguous-k layout used here: lanes 0-15 -> rows 0-15 with K in {0..15};
//             lanes 16-31 -> rows 0-15 with K in {16..31}.
__device__ __forceinline__ void load_a_frag(const __bf16* src, int ld,
                                            bf16x16& a) {
  const int lane = threadIdx.x & 31;
  const int row  = lane & 15;
  const int koff = (lane < 16) ? 0 : 16;
  const __bf16* p = src + row * ld + koff;
#pragma unroll
  for (int i = 0; i < 4; ++i) {
    a[2 * i]         = p[2 * i];
    a[2 * i + 1]     = p[2 * i + 1];
    a[8 + 2 * i]     = p[8 + 2 * i];
    a[8 + 2 * i + 1] = p[8 + 2 * i + 1];
  }
}

// B fragment (32x16 KxN, row-major source, stride ld).
// Layout: VGPR v holds packed K = kb+2v, kb+2v+1; lanes 0-15 -> N, kb=0;
// lanes 16-31 -> N, kb=16.
__device__ __forceinline__ void load_b_frag_rm(const __bf16* src, int ld,
                                               bf16x16& b) {
  const int lane = threadIdx.x & 31;
  const int n  = lane & 15;
  const int kb = (lane < 16) ? 0 : 16;
#pragma unroll
  for (int v = 0; v < 8; ++v) {
    b[2 * v]     = src[(kb + 2 * v) * ld + n];
    b[2 * v + 1] = src[(kb + 2 * v + 1) * ld + n];
  }
}

// B fragment where the source is N-major (keys[n][k]): B(k,n) = src[n*ld + k].
// Per-lane reads are 16 contiguous bf16 -> compiler emits 2x global_load_b128.
__device__ __forceinline__ void load_b_frag_tr(const __bf16* src, int ld,
                                               bf16x16& b) {
  const int lane = threadIdx.x & 31;
  const int n  = lane & 15;
  const int kb = (lane < 16) ? 0 : 16;
  const __bf16* p = src + n * ld + kb;
#pragma unroll
  for (int v = 0; v < 8; ++v) {
    b[2 * v]     = p[2 * v];
    b[2 * v + 1] = p[2 * v + 1];
  }
}

__device__ __forceinline__ float selu_f(float v) {
  const float scale = 1.0507009873554805f;
  const float alpha = 1.6732632423543772f;
  return scale * (v > 0.f ? v : alpha * (expf(v) - 1.f));
}

// ---------------------------------------------------------------------------
// Problem constants (from reference)
// ---------------------------------------------------------------------------
#define BATCH   4096
#define X_DIM   128
#define T_DIM   32
#define IN_DIM  160
#define HID     512
#define KDIM    64
#define NTRAIN  65536
#define TOPK    64

// ---------------------------------------------------------------------------
// Kernel 1: sinusoidal time embedding + concat -> bf16 (B x 160)
// ---------------------------------------------------------------------------
__global__ void build_input_kernel(const float* __restrict__ x,
                                   const float* __restrict__ t,
                                   __bf16* __restrict__ A) {
  int i = blockIdx.x * blockDim.x + threadIdx.x;
  if (i >= BATCH * IN_DIM) return;
  int b = i / IN_DIM, c = i % IN_DIM;
  float v;
  if (c < X_DIM) {
    v = x[b * X_DIM + c];
  } else {
    int j = c - X_DIM;                 // 0..31
    int f = (j < 16) ? j : j - 16;     // frequency index
    float freq = expf(-logf(10000.0f) * (float)f * (1.0f / 16.0f));
    float arg  = t[b] * freq;
    v = (j < 16) ? sinf(arg) : cosf(arg);
  }
  A[i] = (__bf16)v;
}

// ---------------------------------------------------------------------------
// Kernel 2: f32 -> bf16 conversion (keys)
// ---------------------------------------------------------------------------
__global__ void cvt_bf16_kernel(const float* __restrict__ src,
                                __bf16* __restrict__ dst, int n) {
  int i = blockIdx.x * blockDim.x + threadIdx.x;
  if (i < n) dst[i] = (__bf16)src[i];
}

// ---------------------------------------------------------------------------
// Kernel 3: C = act(A @ W + bias), A bf16 (B x KT), W f32 (KT x NT), out bf16.
// Workgroup tile: 128(M) x 64(N); 8 waves in 4x2; each wave: 32x32 (2x2 WMMA
// tiles). K staged through LDS in steps of 32.
// ACT: 0 = SELU, 1 = scale by 0.125 (q / sqrt(KEY_DIM) fold)
// ---------------------------------------------------------------------------
template <int KT, int NT, int ACT>
__global__ __launch_bounds__(256) void gemm_bias_act_kernel(
    const __bf16* __restrict__ A, const float* __restrict__ W,
    const float* __restrict__ bias, __bf16* __restrict__ Out) {
  __shared__ unsigned int a_sh[128][17];   // 128 rows x 32 bf16 (pad 1 dword)
  __shared__ __bf16       w_sh[32][66];    // 32 k x 64 n (pad 2 bf16)

  const int tid   = threadIdx.x;
  const int lane  = tid & 31;
  const int wave  = tid >> 5;
  const int waveM = wave >> 1;             // 0..3
  const int waveN = wave & 1;              // 0..1
  const int mbase = blockIdx.x * 128;
  const int nbase = blockIdx.y * 64;

  f32x8 c[2][2] = {};
  const unsigned int* Ag = (const unsigned int*)A;
  const int ldau = KT / 2;

  for (int k0 = 0; k0 < KT; k0 += 32) {
    // Stage A tile: 128 x 32 bf16 as 128 x 16 dwords
    for (int i = tid; i < 128 * 16; i += 256) {
      int r = i >> 4, kp = i & 15;
      a_sh[r][kp] = Ag[(mbase + r) * ldau + (k0 >> 1) + kp];
    }
    // Stage W tile: 32 x 64, f32 -> bf16
    for (int i = tid; i < 32 * 64; i += 256) {
      int kk = i >> 6, nn = i & 63;
      w_sh[kk][nn] = (__bf16)W[(k0 + kk) * NT + nbase + nn];
    }
    __syncthreads();

    const __bf16* abase = reinterpret_cast<const __bf16*>(&a_sh[0][0]);
    bf16x16 af[2], bfr[2];
    load_a_frag(abase + (waveM * 32 + 0)  * 34, 34, af[0]);
    load_a_frag(abase + (waveM * 32 + 16) * 34, 34, af[1]);
    load_b_frag_rm(&w_sh[0][waveN * 32 + 0],  66, bfr[0]);
    load_b_frag_rm(&w_sh[0][waveN * 32 + 16], 66, bfr[1]);
#pragma unroll
    for (int mt = 0; mt < 2; ++mt)
#pragma unroll
      for (int nt = 0; nt < 2; ++nt)
        c[mt][nt] = wmma_bf16(af[mt], bfr[nt], c[mt][nt]);
    __syncthreads();
  }

  // Epilogue: C/D layout: lanes 0-15 -> N=lane, M=r; lanes 16-31 -> M=8+r.
  const int nlo  = lane & 15;
  const int radd = (lane < 16) ? 0 : 8;
#pragma unroll
  for (int mt = 0; mt < 2; ++mt)
#pragma unroll
    for (int nt = 0; nt < 2; ++nt) {
      int col  = nbase + waveN * 32 + nt * 16 + nlo;
      float bc = bias[col];
#pragma unroll
      for (int r = 0; r < 8; ++r) {
        int row = mbase + waveM * 32 + mt * 16 + radd + r;
        float v = c[mt][nt][r] + bc;
        if (ACT == 0) v = selu_f(v);
        else          v *= 0.125f;   // 1/sqrt(KEY_DIM) folded into q
        Out[row * NT + col] = (__bf16)v;
      }
    }
}

// ---------------------------------------------------------------------------
// Kernel 4: fused logits (q @ keys^T, scale already in q) + streaming top-64
// + softmax. One workgroup per 16 rows of B; 8 waves each produce TWO 16x16
// logits tiles per iteration (256 keys / iteration, 256 iterations).
// B-fragments are double-buffered: the next iteration's key loads are issued
// before the filter/insert barrier phase, hiding L2 latency behind top-k
// bookkeeping.
// ---------------------------------------------------------------------------
__global__ __launch_bounds__(256) void logits_topk_kernel(
    const __bf16* __restrict__ q, const __bf16* __restrict__ keysb,
    float* __restrict__ out) {
  __shared__ float tile[16][260];      // 16 rows x 256 keys (+pad)
  __shared__ float topv[16][TOPK];
  __shared__ int   topi[16][TOPK];
  __shared__ float rowmin[16];
  __shared__ int   rowminpos[16];
  __shared__ float candv[16][256];
  __shared__ int   candi[16][256];
  __shared__ int   candn[16];

  const int tid     = threadIdx.x;
  const int lane    = tid & 31;
  const int wave    = tid >> 5;
  const int rowbase = blockIdx.x * 16;

  for (int i = tid; i < 16 * TOPK; i += 256) {
    topv[i >> 6][i & 63] = -3.0e38f;
    topi[i >> 6][i & 63] = 0;
  }
  if (tid < 16) { rowmin[tid] = -3.0e38f; rowminpos[tid] = 0; candn[tid] = 0; }

  // q fragments for this 16-row tile live in registers for the whole kernel.
  bf16x16 a0, a1;
  load_a_frag(q + rowbase * KDIM,      KDIM, a0);   // K = 0..31
  load_a_frag(q + rowbase * KDIM + 32, KDIM, a1);   // K = 32..63

  // Double-buffered B fragments: [buf][tile0_k0, tile0_k1, tile1_k0, tile1_k1]
  bf16x16 frags[2][4];
  {
    const int kb = wave * 32;
    load_b_frag_tr(keysb + (size_t)kb * KDIM,             KDIM, frags[0][0]);
    load_b_frag_tr(keysb + (size_t)kb * KDIM + 32,        KDIM, frags[0][1]);
    load_b_frag_tr(keysb + (size_t)(kb + 16) * KDIM,      KDIM, frags[0][2]);
    load_b_frag_tr(keysb + (size_t)(kb + 16) * KDIM + 32, KDIM, frags[0][3]);
  }
  __syncthreads();

  const int nlo  = lane & 15;
  const int radd = (lane < 16) ? 0 : 8;
  const int frow = tid >> 4;           // filter row (0..15)
  const int fcol = (tid & 15) * 16;    // filter col base (0..240)

  constexpr int NITER = NTRAIN / 256;  // 256
#pragma unroll 2
  for (int it = 0; it < NITER; ++it) {
    const bf16x16* cur = frags[it & 1];
    bf16x16*       nxt = frags[(it + 1) & 1];

    // Two 16x16 logits tiles for this wave (K = 64 -> 2 WMMAs each).
    f32x8 c0 = {}, c1 = {};
    c0 = wmma_bf16(a0, cur[0], c0);
    c0 = wmma_bf16(a1, cur[1], c0);
    c1 = wmma_bf16(a0, cur[2], c1);
    c1 = wmma_bf16(a1, cur[3], c1);

    const int col0 = wave * 32 + nlo;
#pragma unroll
    for (int r = 0; r < 8; ++r) {
      tile[radd + r][col0]      = c0[r];
      tile[radd + r][col0 + 16] = c1[r];
    }

    // Prefetch next iteration's keys; latency hidden by filter/insert below.
    {
      const int nit = (it + 1 < NITER) ? it + 1 : it;
      const int kb  = nit * 256 + wave * 32;
      load_b_frag_tr(keysb + (size_t)kb * KDIM,             KDIM, nxt[0]);
      load_b_frag_tr(keysb + (size_t)kb * KDIM + 32,        KDIM, nxt[1]);
      load_b_frag_tr(keysb + (size_t)(kb + 16) * KDIM,      KDIM, nxt[2]);
      load_b_frag_tr(keysb + (size_t)(kb + 16) * KDIM + 32, KDIM, nxt[3]);
    }
    __syncthreads();

    // Parallel threshold filter: each thread scans 16 contiguous values of
    // one row (vectorized LDS reads). Stale rowmin is conservative->correct.
    {
      const float th = rowmin[frow];
      const float4* tp = (const float4*)&tile[frow][fcol];
#pragma unroll
      for (int v4 = 0; v4 < 4; ++v4) {
        float4 vv = tp[v4];
        float vs[4] = {vv.x, vv.y, vv.z, vv.w};
#pragma unroll
        for (int j = 0; j < 4; ++j) {
          if (vs[j] > th) {
            int p = atomicAdd(&candn[frow], 1);
            candv[frow][p] = vs[j];
            candi[frow][p] = it * 256 + fcol + v4 * 4 + j;
          }
        }
      }
    }
    __syncthreads();

    // Owner lanes: serial replace-min insert (rare after warm-up).
    if (tid < 16) {
      int r  = tid;
      int nc = candn[r];
      for (int cd = 0; cd < nc; ++cd) {
        float v = candv[r][cd];
        if (v > rowmin[r]) {
          int p = rowminpos[r];
          topv[r][p] = v;
          topi[r][p] = candi[r][cd];
          float mn = topv[r][0]; int mp = 0;
          for (int j = 1; j < TOPK; ++j) {
            float tv = topv[r][j];
            if (tv < mn) { mn = tv; mp = j; }
          }
          rowmin[r] = mn; rowminpos[r] = mp;
        }
      }
      candn[r] = 0;
    }
    __syncthreads();
  }

  // Final: sort descending, softmax, write weights then indices (as f32).
  if (tid < 16) {
    int r = tid;
    for (int i = 0; i < TOPK - 1; ++i) {
      int best = i; float bv = topv[r][i];
      for (int j = i + 1; j < TOPK; ++j)
        if (topv[r][j] > bv) { bv = topv[r][j]; best = j; }
      if (best != i) {
        float tv = topv[r][i]; topv[r][i] = topv[r][best]; topv[r][best] = tv;
        int   ti = topi[r][i]; topi[r][i] = topi[r][best]; topi[r][best] = ti;
      }
    }
    float m = topv[r][0];
    float s = 0.f;
    float e[TOPK];
    for (int j = 0; j < TOPK; ++j) { e[j] = expf(topv[r][j] - m); s += e[j]; }
    float inv = 1.f / s;
    float* ow = out + (size_t)(rowbase + r) * TOPK;
    float* oi = out + (size_t)BATCH * TOPK + (size_t)(rowbase + r) * TOPK;
    for (int j = 0; j < TOPK; ++j) {
      ow[j] = e[j] * inv;
      oi[j] = (float)topi[r][j];
    }
  }
}

// ---------------------------------------------------------------------------
// Launch
// ---------------------------------------------------------------------------
extern "C" void kernel_launch(void* const* d_in, const int* in_sizes, int n_in,
                              void* d_out, int out_size, void* d_ws,
                              size_t ws_size, hipStream_t stream) {
  const float* x    = (const float*)d_in[0];
  const float* t    = (const float*)d_in[1];
  const float* W0   = (const float*)d_in[2];
  const float* b0   = (const float*)d_in[3];
  const float* W1   = (const float*)d_in[4];
  const float* b1   = (const float*)d_in[5];
  const float* W2   = (const float*)d_in[6];
  const float* b2   = (const float*)d_in[7];
  const float* Wq   = (const float*)d_in[8];
  const float* bq   = (const float*)d_in[9];
  const float* keys = (const float*)d_in[10];

  char* ws = (char*)d_ws;
  size_t off = 0;
  auto alloc = [&](size_t bytes) -> void* {
    void* p = ws + off;
    off = (off + bytes + 255) & ~(size_t)255;
    return p;
  };
  __bf16* Ain   = (__bf16*)alloc((size_t)BATCH * IN_DIM * 2);
  __bf16* h0    = (__bf16*)alloc((size_t)BATCH * HID * 2);
  __bf16* h1    = (__bf16*)alloc((size_t)BATCH * HID * 2);
  __bf16* h2    = (__bf16*)alloc((size_t)BATCH * HID * 2);
  __bf16* qbf   = (__bf16*)alloc((size_t)BATCH * KDIM * 2);
  __bf16* keysb = (__bf16*)alloc((size_t)NTRAIN * KDIM * 2);

  // 1) time embedding + concat
  build_input_kernel<<<(BATCH * IN_DIM + 255) / 256, 256, 0, stream>>>(x, t, Ain);
  // 2) keys -> bf16
  cvt_bf16_kernel<<<(NTRAIN * KDIM + 255) / 256, 256, 0, stream>>>(
      keys, keysb, NTRAIN * KDIM);
  // 3) MLP (SELU) + q head (x0.125 logit-scale fold)
  gemm_bias_act_kernel<IN_DIM, HID, 0>
      <<<dim3(BATCH / 128, HID / 64), 256, 0, stream>>>(Ain, W0, b0, h0);
  gemm_bias_act_kernel<HID, HID, 0>
      <<<dim3(BATCH / 128, HID / 64), 256, 0, stream>>>(h0, W1, b1, h1);
  gemm_bias_act_kernel<HID, HID, 0>
      <<<dim3(BATCH / 128, HID / 64), 256, 0, stream>>>(h1, W2, b2, h2);
  gemm_bias_act_kernel<HID, KDIM, 1>
      <<<dim3(BATCH / 128, KDIM / 64), 256, 0, stream>>>(h2, Wq, bq, qbf);
  // 4) fused logits + top-64 + softmax
  logits_topk_kernel<<<BATCH / 16, 256, 0, stream>>>(qbf, keysb, (float*)d_out);
}
